// ReadSetClassifier_54065048322165
// MI455X (gfx1250) — matrix operand order repaired and checked by reading, not compiled, not run on hardware
//
#include <hip/hip_runtime.h>
#include <hip/hip_bf16.h>
#include <cmath>

// ---------------------------------------------------------------------------
// ReadSetClassifier for MI455X (gfx1250): fused Deep-Sets forward pass.
// v_wmma_f32_16x16x32_f16 with inline-zero C accumulator (bias folded into
// the epilogue); weights pre-swizzled to fragment layout in LDS and hoisted
// into VGPRs once per wave (phi); activations ping-pong through LDS in
// fragment layout so every WMMA A-operand is one 32B vector load.
// ---------------------------------------------------------------------------

typedef _Float16 h16;
typedef __attribute__((ext_vector_type(2)))  _Float16 h2;
typedef __attribute__((ext_vector_type(16))) _Float16 v16h;
typedef __attribute__((ext_vector_type(8)))  float    v8f;

__device__ __forceinline__ h2 pkrtz(float a, float b) {
    return __builtin_bit_cast(h2, __builtin_amdgcn_cvt_pkrtz(a, b));
}

__device__ __forceinline__ void lds_fence() {
    asm volatile("s_wait_dscnt 0" ::: "memory");
}

// D = A*B (C = inline 0, no register cost)
__device__ __forceinline__ v8f wmma_z(v16h a, v16h b) {
    v8f z = {0.f, 0.f, 0.f, 0.f, 0.f, 0.f, 0.f, 0.f};
    return __builtin_amdgcn_wmma_f32_16x16x32_f16(
        false, a, false, b, (short)0, z, false, false);
}
// D = A*B + C
__device__ __forceinline__ v8f wmma_c(v16h a, v16h b, v8f c) {
    return __builtin_amdgcn_wmma_f32_16x16x32_f16(
        false, a, false, b, (short)0, c, false, false);
}

__device__ __forceinline__ float reluf_(float x) { return x > 0.f ? x : 0.f; }

// decompose a K-index kk (0..31 within a chunk) into (half, e)
__device__ __forceinline__ void kdecomp(int kk, int& halfp, int& ep) {
    halfp = (kk >> 3) & 1;
    ep    = (kk & 7) + ((kk >> 4) << 3);
}

// vector load of one fragment (A or B) stored fragment-contiguous in LDS
__device__ __forceinline__ v16h load_frag(const h16* base, int fragIdx, int lane) {
    return *(const v16h*)(base + (fragIdx << 9) + (lane << 4));
}

// First-layer A tile from global f32 rows (32 feats each), rows >= valid -> 0
__device__ __forceinline__ v16h load_a_global(const float* __restrict__ src,
                                              long row0, int valid, int lane) {
    const int half = lane >> 4, m = lane & 15;
    v16h a;
    if (m < valid) {
        const float* row = src + (size_t)(row0 + m) * 32;
        const float4 f0 = ((const float4*)(row + half * 8))[0];
        const float4 f1 = ((const float4*)(row + half * 8))[1];
        const float4 f2 = ((const float4*)(row + 16 + half * 8))[0];
        const float4 f3 = ((const float4*)(row + 16 + half * 8))[1];
        h2 p0 = pkrtz(f0.x, f0.y);
        h2 p1 = pkrtz(f0.z, f0.w);
        h2 p2 = pkrtz(f1.x, f1.y);
        h2 p3 = pkrtz(f1.z, f1.w);
        h2 p4 = pkrtz(f2.x, f2.y);
        h2 p5 = pkrtz(f2.z, f2.w);
        h2 p6 = pkrtz(f3.x, f3.y);
        h2 p7 = pkrtz(f3.z, f3.w);
        a[0]=p0[0]; a[1]=p0[1]; a[2]=p1[0]; a[3]=p1[1];
        a[4]=p2[0]; a[5]=p2[1]; a[6]=p3[0]; a[7]=p3[1];
        a[8]=p4[0]; a[9]=p4[1]; a[10]=p5[0]; a[11]=p5[1];
        a[12]=p6[0]; a[13]=p6[1]; a[14]=p7[0]; a[15]=p7[1];
    } else {
#pragma unroll
        for (int e = 0; e < 16; ++e) a[e] = (h16)0.f;
    }
    return a;
}

// Store D tile (rows m=r+half*8, feature kp=ncol0+n) into fragment-layout act
// applying +bias then ReLU.  addr = cc*512 + halfp*256 + m*16 + ep
__device__ __forceinline__ void store_d_brelu_frag(h16* act, int ncol0,
                                                   v8f d, float bv, int lane) {
    const int half = lane >> 4, n = lane & 15;
    const int kp = ncol0 + n;
    const int cc = kp >> 5;
    int halfp, ep; kdecomp(kp & 31, halfp, ep);
    h16* base = act + (cc << 9) + (halfp << 8) + ep;
#pragma unroll
    for (int r = 0; r < 8; r += 2) {
        h2 p = pkrtz(reluf_(d[r] + bv), reluf_(d[r + 1] + bv));
        base[(r + half * 8) << 4]     = p[0];
        base[(r + 1 + half * 8) << 4] = p[1];
    }
}

__device__ __forceinline__ float sigmoidf_(float x) {
    return 1.f / (1.f + __expf(-x));
}

__device__ __forceinline__ void store_d_bsig_frag(h16* act, int ncol0,
                                                  v8f d, float bv, int lane) {
    const int half = lane >> 4, n = lane & 15;
    const int kp = ncol0 + n;
    const int cc = kp >> 5;
    int halfp, ep; kdecomp(kp & 31, halfp, ep);
    h16* base = act + (cc << 9) + (halfp << 8) + ep;
#pragma unroll
    for (int r = 0; r < 8; r += 2) {
        h2 p = pkrtz(sigmoidf_(d[r] + bv), sigmoidf_(d[r + 1] + bv));
        base[(r + half * 8) << 4]     = p[0];
        base[(r + 1 + half * 8) << 4] = p[1];
    }
}

// Store D tile into a plain row-major [16][ld] f16 buffer (for scalar use)
__device__ __forceinline__ void store_d_brelu_rm(h16* act, int ld, int ncol0,
                                                 v8f d, float bv, int lane) {
    const int half = lane >> 4, n = lane & 15;
#pragma unroll
    for (int r = 0; r < 8; r += 2) {
        h2 p = pkrtz(reluf_(d[r] + bv), reluf_(d[r + 1] + bv));
        act[(r + half * 8) * ld + ncol0 + n]     = p[0];
        act[(r + 1 + half * 8) * ld + ncol0 + n] = p[1];
    }
}

// Stage a [K][N] f32 weight into fragment-contiguous f16 LDS.
__device__ __forceinline__ void stage_weight_swz(const float* __restrict__ W,
                                                 h16* swz, int K, int N,
                                                 int tid, int nthreads) {
    const int ncn = N >> 4;
    const int total = K * N;
    for (int idx = tid; idx < total; idx += nthreads) {
        const int f = idx >> 9;
        const int r = idx & 511;
        const int lane = r >> 4, e = r & 15;
        const int half = lane >> 4, n = lane & 15;
        const int kc = f / ncn, nc = f - kc * ncn;
        const int k = kc * 32 + half * 8 + (e & 7) + ((e >> 3) << 4);
        swz[idx] = (h16)W[k * N + nc * 16 + n];
    }
}

// ---------------------------------------------------------------------------
// Kernel 1: exclusive prefix scan of segment counts (one 256-thread block)
// ---------------------------------------------------------------------------
__global__ void k_scan(const int* __restrict__ counts, int* __restrict__ offs, int n) {
    __shared__ int ss[256];
    const int t = threadIdx.x;
    const int per = n / 256;
    const int base = t * per;
    int s = 0;
    for (int i = 0; i < per; ++i) s += counts[base + i];
    ss[t] = s;
    __syncthreads();
    for (int off = 1; off < 256; off <<= 1) {
        int v = (t >= off) ? ss[t - off] : 0;
        __syncthreads();
        ss[t] += v;
        __syncthreads();
    }
    int run = ss[t] - s;
    for (int i = 0; i < per; ++i) { offs[base + i] = run; run += counts[base + i]; }
}

// ---------------------------------------------------------------------------
// Kernel 2: phi MLP (32->64->64->32, sigmoid) + fused segment mean.
// Weight fragments hoisted into registers once per wave; biases are scalars.
// ---------------------------------------------------------------------------
struct PhiW {
    v16h w1[4];          // layer1: nc 0..3
    v16h w2[8];          // layer2: kc*4 + nc
    v16h w3[4];          // layer3: kc*2 + nc
    float b1[4], b2[4], b3[2];   // per-lane bias values (scalars)
};

__device__ __forceinline__ void phi_segment(
    const float* __restrict__ src, long start, int count,
    h16* actA, h16* actB, const PhiW& w,
    float* __restrict__ means, int s, int lane) {

    const int half = lane >> 4;
    float acc0 = 0.f, acc1 = 0.f;
    const int ntiles = (count + 15) >> 4;

    for (int t = 0; t < ntiles; ++t) {
        const int valid = min(16, count - t * 16);
        // layer 1: 32 -> 64
        v16h a = load_a_global(src, start + (long)t * 16, valid, lane);
#pragma unroll
        for (int nt = 0; nt < 4; ++nt) {
            v8f d = wmma_z(a, w.w1[nt]);
            store_d_brelu_frag(actA, nt * 16, d, w.b1[nt], lane);
        }
        lds_fence();
        // layer 2: 64 -> 64
        v16h a20 = load_frag(actA, 0, lane);
        v16h a21 = load_frag(actA, 1, lane);
#pragma unroll
        for (int nt = 0; nt < 4; ++nt) {
            v8f d = wmma_z(a20, w.w2[0 * 4 + nt]);
            d = wmma_c(a21, w.w2[1 * 4 + nt], d);
            store_d_brelu_frag(actB, nt * 16, d, w.b2[nt], lane);
        }
        lds_fence();
        // layer 3: 64 -> 32, sigmoid, masked row-accumulate
        v16h a30 = load_frag(actB, 0, lane);
        v16h a31 = load_frag(actB, 1, lane);
#pragma unroll
        for (int nt = 0; nt < 2; ++nt) {
            v8f d = wmma_z(a30, w.w3[0 * 2 + nt]);
            d = wmma_c(a31, w.w3[1 * 2 + nt], d);
            float part = 0.f;
#pragma unroll
            for (int r = 0; r < 8; ++r) {
                float v = sigmoidf_(d[r] + w.b3[nt]);
                part += ((r + half * 8) < valid) ? v : 0.f;
            }
            if (nt == 0) acc0 += part; else acc1 += part;
        }
    }
    acc0 += __shfl_xor(acc0, 16, 32);
    acc1 += __shfl_xor(acc1, 16, 32);
    const float inv = 1.f / (float)count;
    if (lane < 16) {
        means[(long)s * 32 + lane]      = acc0 * inv;
        means[(long)s * 32 + 16 + lane] = acc1 * inv;
    }
}

__global__ __launch_bounds__(128) void k_phi(
    const float* __restrict__ ref, const float* __restrict__ alt,
    const float* __restrict__ W1, const float* __restrict__ B1,
    const float* __restrict__ W2, const float* __restrict__ B2,
    const float* __restrict__ W3, const float* __restrict__ B3,
    const int* __restrict__ ref_counts, const int* __restrict__ alt_counts,
    const int* __restrict__ ref_off, const int* __restrict__ alt_off,
    float* __restrict__ ref_means, float* __restrict__ alt_means,
    int num_sets) {

    __shared__ __align__(32) h16 sW1[32 * 64];
    __shared__ __align__(32) h16 sW2[64 * 64];
    __shared__ __align__(32) h16 sW3[64 * 32];
    __shared__ float sB1[64], sB2[64], sB3[32];
    __shared__ __align__(32) h16 sAct[4][2][16 * 64];

    const int tid = threadIdx.x;
    stage_weight_swz(W1, sW1, 32, 64, tid, 128);
    stage_weight_swz(W2, sW2, 64, 64, tid, 128);
    stage_weight_swz(W3, sW3, 64, 32, tid, 128);
    if (tid < 64) { sB1[tid] = B1[tid]; sB2[tid] = B2[tid]; }
    if (tid < 32) sB3[tid] = B3[tid];
    __syncthreads();

    const int wave = tid >> 5, lane = tid & 31;
    const int s = blockIdx.x * 4 + wave;
    if (s >= num_sets) return;   // wave-uniform

    PhiW w;
#pragma unroll
    for (int f = 0; f < 4; ++f) w.w1[f] = load_frag(sW1, f, lane);
#pragma unroll
    for (int f = 0; f < 8; ++f) w.w2[f] = load_frag(sW2, f, lane);
#pragma unroll
    for (int f = 0; f < 4; ++f) w.w3[f] = load_frag(sW3, f, lane);
    const int n = lane & 15;
#pragma unroll
    for (int nt = 0; nt < 4; ++nt) { w.b1[nt] = sB1[nt * 16 + n]; w.b2[nt] = sB2[nt * 16 + n]; }
#pragma unroll
    for (int nt = 0; nt < 2; ++nt) w.b3[nt] = sB3[nt * 16 + n];

    h16* actA = &sAct[wave][0][0];
    h16* actB = &sAct[wave][1][0];

    phi_segment(ref, (long)ref_off[s], ref_counts[s], actA, actB, w,
                ref_means, s, lane);
    phi_segment(alt, (long)alt_off[s], alt_counts[s], actA, actB, w,
                alt_means, s, lane);
}

// ---------------------------------------------------------------------------
// Kernel 3: omega (32->64->32, sigmoid) + rho (96->64->32->1) per 16-set tile.
// Single pass per wave -> weight fragments stay in LDS (no reuse to exploit).
// ---------------------------------------------------------------------------
__global__ __launch_bounds__(128) void k_rho(
    const float* __restrict__ info,
    const float* __restrict__ oW1, const float* __restrict__ oB1,
    const float* __restrict__ oW2, const float* __restrict__ oB2,
    const float* __restrict__ rW1, const float* __restrict__ rB1,
    const float* __restrict__ rW2, const float* __restrict__ rB2,
    const float* __restrict__ rW3, const float* __restrict__ rB3,
    const float* __restrict__ ref_means, const float* __restrict__ alt_means,
    const int* __restrict__ alt_counts,
    float* __restrict__ out, int num_sets) {

    __shared__ __align__(32) h16 sOW1[32 * 64];   // 4 frags
    __shared__ __align__(32) h16 sOW2[64 * 32];   // f = kc*2+nc
    __shared__ __align__(32) h16 sRW1[96 * 64];   // f = kc*4+nc (kc 0..2)
    __shared__ __align__(32) h16 sRW2[64 * 32];   // f = kc*2+nc
    __shared__ float sOB1[64], sOB2[32], sRB1[64], sRB2[32], sRW3f[32];
    __shared__ __align__(32) h16 sStack[4][3 * 512];   // 16x96 fragment layout
    __shared__ __align__(32) h16 sH1[4][2 * 512];      // 16x64 fragment layout
    __shared__ __align__(32) h16 sH2[4][16 * 32];      // row-major for final dot

    const int tid = threadIdx.x;
    stage_weight_swz(oW1, sOW1, 32, 64, tid, 128);
    stage_weight_swz(oW2, sOW2, 64, 32, tid, 128);
    stage_weight_swz(rW1, sRW1, 96, 64, tid, 128);
    stage_weight_swz(rW2, sRW2, 64, 32, tid, 128);
    if (tid < 64) { sOB1[tid] = oB1[tid]; sRB1[tid] = rB1[tid]; }
    if (tid < 32) { sOB2[tid] = oB2[tid]; sRB2[tid] = rB2[tid]; sRW3f[tid] = rW3[tid]; }
    __syncthreads();

    const int wave = tid >> 5, lane = tid & 31;
    const int n = lane & 15;
    const int ts = (blockIdx.x * 4 + wave) * 16;
    if (ts >= num_sets) return;   // wave-uniform

    h16* stk = &sStack[wave][0];
    h16* h1  = &sH1[wave][0];
    h16* h2m = &sH2[wave][0];

    // per-lane biases into scalar registers
    float ob1[4], ob2[2], rb1[4], rb2[2];
#pragma unroll
    for (int nt = 0; nt < 4; ++nt) { ob1[nt] = sOB1[nt * 16 + n]; rb1[nt] = sRB1[nt * 16 + n]; }
#pragma unroll
    for (int nt = 0; nt < 2; ++nt) { ob2[nt] = sOB2[nt * 16 + n]; rb2[nt] = sRB2[nt * 16 + n]; }

    // stacked chunks 0-1 <- [ref_means | alt_means], fragment layout scatter
    for (int i = lane; i < 16 * 16; i += 32) {
        const int m = i >> 4, q = i & 15;
        const float4 v = (q < 8)
            ? ((const float4*)(ref_means + (size_t)(ts + m) * 32))[q]
            : ((const float4*)(alt_means + (size_t)(ts + m) * 32))[q - 8];
        const float vv[4] = {v.x, v.y, v.z, v.w};
#pragma unroll
        for (int j = 0; j < 4; ++j) {
            const int kp = q * 4 + j;           // 0..63
            const int cc = kp >> 5;
            int halfp, ep; kdecomp(kp & 31, halfp, ep);
            stk[(cc << 9) + (halfp << 8) + (m << 4) + ep] = (h16)vv[j];
        }
    }

    // omega layer 1: 32 -> 64
    v16h a = load_a_global(info, ts, 16, lane);
#pragma unroll
    for (int nt = 0; nt < 4; ++nt) {
        v8f d = wmma_z(a, load_frag(sOW1, nt, lane));
        store_d_brelu_frag(h1, nt * 16, d, ob1[nt], lane);
    }
    lds_fence();
    // omega layer 2: 64 -> 32, sigmoid -> stacked chunk 2 (features 64..95)
    {
        v16h a0 = load_frag(h1, 0, lane);
        v16h a1 = load_frag(h1, 1, lane);
#pragma unroll
        for (int nt = 0; nt < 2; ++nt) {
            v8f d = wmma_z(a0, load_frag(sOW2, 0 * 2 + nt, lane));
            d = wmma_c(a1, load_frag(sOW2, 1 * 2 + nt, lane), d);
            store_d_bsig_frag(stk, 64 + nt * 16, d, ob2[nt], lane);
        }
    }
    lds_fence();
    // rho layer 1: 96 -> 64 (3 K-chunks)
    {
        v16h s0 = load_frag(stk, 0, lane);
        v16h s1 = load_frag(stk, 1, lane);
        v16h s2 = load_frag(stk, 2, lane);
#pragma unroll
        for (int nt = 0; nt < 4; ++nt) {
            v8f d = wmma_z(s0, load_frag(sRW1, 0 * 4 + nt, lane));
            d = wmma_c(s1, load_frag(sRW1, 1 * 4 + nt, lane), d);
            d = wmma_c(s2, load_frag(sRW1, 2 * 4 + nt, lane), d);
            store_d_brelu_frag(h1, nt * 16, d, rb1[nt], lane);
        }
    }
    lds_fence();
    // rho layer 2: 64 -> 32 (row-major output for the scalar final layer)
    {
        v16h a0 = load_frag(h1, 0, lane);
        v16h a1 = load_frag(h1, 1, lane);
#pragma unroll
        for (int nt = 0; nt < 2; ++nt) {
            v8f d = wmma_z(a0, load_frag(sRW2, 0 * 2 + nt, lane));
            d = wmma_c(a1, load_frag(sRW2, 1 * 2 + nt, lane), d);
            store_d_brelu_rm(h2m, 32, nt * 16, d, rb2[nt], lane);
        }
    }
    lds_fence();
    // rho layer 3: 32 -> 1, scale by sqrt(alt_count)
    if (lane < 16) {
        const int sidx = ts + lane;
        float dot = rB3[0];
#pragma unroll
        for (int j = 0; j < 32; ++j)
            dot += (float)h2m[lane * 32 + j] * sRW3f[j];
        out[sidx] = dot * sqrtf((float)alt_counts[sidx]);
    }
}

// ---------------------------------------------------------------------------
// Host-side launcher
// ---------------------------------------------------------------------------
extern "C" void kernel_launch(void* const* d_in, const int* in_sizes, int n_in,
                              void* d_out, int out_size, void* d_ws, size_t ws_size,
                              hipStream_t stream) {
    const float* ref  = (const float*)d_in[0];
    const float* alt  = (const float*)d_in[1];
    const float* info = (const float*)d_in[2];
    const float* pW1 = (const float*)d_in[3];  const float* pB1 = (const float*)d_in[4];
    const float* pW2 = (const float*)d_in[5];  const float* pB2 = (const float*)d_in[6];
    const float* pW3 = (const float*)d_in[7];  const float* pB3 = (const float*)d_in[8];
    const float* oW1 = (const float*)d_in[9];  const float* oB1 = (const float*)d_in[10];
    const float* oW2 = (const float*)d_in[11]; const float* oB2 = (const float*)d_in[12];
    const float* rW1 = (const float*)d_in[13]; const float* rB1 = (const float*)d_in[14];
    const float* rW2 = (const float*)d_in[15]; const float* rB2 = (const float*)d_in[16];
    const float* rW3 = (const float*)d_in[17]; const float* rB3 = (const float*)d_in[18];
    const int* ref_counts = (const int*)d_in[19];
    const int* alt_counts = (const int*)d_in[20];

    const int num_sets = in_sizes[2] / 32;   // 16384

    int*   ref_off   = (int*)d_ws;
    int*   alt_off   = ref_off + num_sets;
    float* ref_means = (float*)(alt_off + num_sets);
    float* alt_means = ref_means + (size_t)num_sets * 32;
    float* out       = (float*)d_out;

    k_scan<<<1, 256, 0, stream>>>(ref_counts, ref_off, num_sets);
    k_scan<<<1, 256, 0, stream>>>(alt_counts, alt_off, num_sets);

    const int phi_blocks = (num_sets + 3) / 4;
    k_phi<<<phi_blocks, 128, 0, stream>>>(
        ref, alt, pW1, pB1, pW2, pB2, pW3, pB3,
        ref_counts, alt_counts, ref_off, alt_off,
        ref_means, alt_means, num_sets);

    const int tiles = (num_sets + 15) / 16;
    const int rho_blocks = (tiles + 3) / 4;
    k_rho<<<rho_blocks, 128, 0, stream>>>(
        info, oW1, oB1, oW2, oB2,
        rW1, rB1, rW2, rB2, rW3, rB3,
        ref_means, alt_means, alt_counts, out, num_sets);
}
